// Decoder_35725537968704
// MI455X (gfx1250) — compile-verified, matrix operand order checked by reading, not run on hardware
//
#include <hip/hip_runtime.h>
#include <hip/hip_bf16.h>

// ---------------------------------------------------------------------------
// Persistent 2-layer LSTM rollout for MI455X (gfx1250, wave32, WMMA).
//   B=64, H=1024, T=1024, L=2.  GEMMs in v_wmma_f32_16x16x32_bf16 (fp32 acc).
//   Weight B-fragments are DMA'd global->LDS with GLOBAL_LOAD_ASYNC_TO_LDS_B128
//   (double-buffered 4-kt chunks, ASYNCcnt-tracked) to kill the 4x L2 fetch
//   redundancy; A-fragments load directly from L2.  Two grid barriers / step.
// ---------------------------------------------------------------------------

typedef __attribute__((ext_vector_type(16))) __bf16 v16bf;
typedef __attribute__((ext_vector_type(8)))  __bf16 v8bf;
typedef __attribute__((ext_vector_type(8)))  float  v8f;

#define BATCH 64
#define HDIM  1024
#define TSTEPS 1024
#define NBLK  64
#define TPB   256

// ---------------- workspace layout (bytes) ----------------
#define WPK0_ELEMS (4096u * 1024u)            // 4H x H   bf16
#define WPK1_ELEMS (4096u * 2048u)            // 4H x 2H  bf16
#define OFF_WPK0   0ull
#define OFF_WPK1   (OFF_WPK0 + (size_t)WPK0_ELEMS * 2)
#define OFF_H0BF   (OFF_WPK1 + (size_t)WPK1_ELEMS * 2)      // 2 x 64x1024 bf16
#define OFF_H1BF   (OFF_H0BF + 2ull * BATCH * HDIM * 2)
#define OFF_C0     (OFF_H1BF + 2ull * BATCH * HDIM * 2)     // 64x1024 f32
#define OFF_C1     (OFF_C0 + (size_t)BATCH * HDIM * 4)
#define OFF_BIAS0  (OFF_C1 + (size_t)BATCH * HDIM * 4)      // 4096 f32
#define OFF_BIAS1  (OFF_BIAS0 + 4096ull * 4)
#define OFF_WIH0   (OFF_BIAS1 + 4096ull * 4)                // 4096 f32
#define OFF_XIN    (OFF_WIH0 + 4096ull * 4)                 // 2 x 64 f32
#define OFF_SYNC   (OFF_XIN + 2ull * 64 * 4)                // 2 u32

__device__ __forceinline__ float sigf(float x) {
  return __builtin_amdgcn_rcpf(1.0f + __expf(-x));
}

#if __has_builtin(__builtin_amdgcn_tanhf)
__device__ __forceinline__ float tanh_fast(float x) { return __builtin_amdgcn_tanhf(x); }
#else
__device__ __forceinline__ float tanh_fast(float x) {
  // branch-free: tanh(x) = 2*sigmoid(2x) - 1   (v_exp_f32 + v_rcp_f32)
  return 2.0f * __builtin_amdgcn_rcpf(1.0f + __expf(-2.0f * x)) - 1.0f;
}
#endif

// ---- CDNA5 async global->LDS copy (ASYNCcnt-tracked), 16B per lane ----
__device__ __forceinline__ void stage_issue(const __bf16* gsrc, const __bf16* lds) {
  unsigned ldsoff = (unsigned)(uintptr_t)lds;          // low 32 bits = LDS offset
  unsigned long long ga = (unsigned long long)(uintptr_t)gsrc;
  asm volatile("global_load_async_to_lds_b128 %0, %1, off"
               :: "v"(ldsoff), "v"(ga) : "memory");
}
__device__ __forceinline__ void async_wait0() {
  asm volatile("s_wait_asynccnt 0" ::: "memory");
}

__device__ __forceinline__ void gsync(unsigned* cnt, unsigned* gen) {
  __syncthreads();
  if (threadIdx.x == 0) {
    __threadfence();
    unsigned g = __hip_atomic_load(gen, __ATOMIC_ACQUIRE, __HIP_MEMORY_SCOPE_AGENT);
    unsigned a = __hip_atomic_fetch_add(cnt, 1u, __ATOMIC_ACQ_REL, __HIP_MEMORY_SCOPE_AGENT);
    if (a == NBLK - 1) {
      __hip_atomic_store(cnt, 0u, __ATOMIC_RELAXED, __HIP_MEMORY_SCOPE_AGENT);
      __threadfence();
      __hip_atomic_fetch_add(gen, 1u, __ATOMIC_ACQ_REL, __HIP_MEMORY_SCOPE_AGENT);
    } else {
      while (__hip_atomic_load(gen, __ATOMIC_ACQUIRE, __HIP_MEMORY_SCOPE_AGENT) == g) {
        __builtin_amdgcn_s_sleep(2);
      }
    }
    __threadfence();
  }
  __syncthreads();
}

// ----- pack W_hh0^T (K=1024, N=4096) into WMMA B-fragments (bf16) -----
__global__ void pack_w0(const float* __restrict__ W, __bf16* __restrict__ out) {
  size_t id = (size_t)blockIdx.x * blockDim.x + threadIdx.x;
  if (id >= (size_t)WPK0_ELEMS) return;
  unsigned within = (unsigned)(id & 511u);
  unsigned frag   = (unsigned)(id >> 9);
  unsigned lane = within >> 4, j = within & 15u;
  unsigned kt = frag & 31u, nt = frag >> 5;
  unsigned n = nt * 16u + (lane & 15u);
  unsigned k = kt * 32u + ((lane >> 4) << 3) + (j < 8u ? j : j + 8u);
  out[id] = (__bf16)W[(size_t)n * HDIM + k];   // W^T[k][n] = W[n][k]
}

// ----- pack [W_ih1^T ; W_hh1^T] (K=2048, N=4096) -----
__global__ void pack_w1(const float* __restrict__ Wih, const float* __restrict__ Whh,
                        __bf16* __restrict__ out) {
  size_t id = (size_t)blockIdx.x * blockDim.x + threadIdx.x;
  if (id >= (size_t)WPK1_ELEMS) return;
  unsigned within = (unsigned)(id & 511u);
  unsigned frag   = (unsigned)(id >> 9);
  unsigned lane = within >> 4, j = within & 15u;
  unsigned kt = frag & 63u, nt = frag >> 6;
  unsigned n = nt * 16u + (lane & 15u);
  unsigned k = kt * 32u + ((lane >> 4) << 3) + (j < 8u ? j : j + 8u);
  float v = (k < HDIM) ? Wih[(size_t)n * HDIM + k]
                       : Whh[(size_t)n * HDIM + (k - HDIM)];
  out[id] = (__bf16)v;
}

__global__ void init_state(const float* __restrict__ z,
                           const float* __restrict__ b_ih0, const float* __restrict__ b_hh0,
                           const float* __restrict__ b_ih1, const float* __restrict__ b_hh1,
                           const float* __restrict__ W_ih0,
                           float* __restrict__ c0, float* __restrict__ c1,
                           __bf16* __restrict__ h0bf, __bf16* __restrict__ h1bf,
                           float* __restrict__ bias0, float* __restrict__ bias1,
                           float* __restrict__ wih0c, float* __restrict__ xin,
                           unsigned* __restrict__ syncb) {
  unsigned tid = blockIdx.x * blockDim.x + threadIdx.x;
  if (tid < BATCH * HDIM) {
    float zv = z[tid];
    c0[tid] = zv;                      // reference inits BOTH h and c to z
    c1[tid] = zv;
    h0bf[BATCH * HDIM + tid] = (__bf16)zv;   // parity-1 buffer = "prev" at t=0
    h1bf[BATCH * HDIM + tid] = (__bf16)zv;
  }
  if (tid < 4096) {
    bias0[tid] = b_ih0[tid] + b_hh0[tid];
    bias1[tid] = b_ih1[tid] + b_hh1[tid];
    wih0c[tid] = W_ih0[tid];
  }
  if (tid < 128) xin[tid] = 0.0f;      // initial feedback input = 0
  if (tid == 0) { syncb[0] = 0u; syncb[1] = 0u; }
}

__global__ void __launch_bounds__(TPB, 2)
lstm_persistent(const float* __restrict__ inputs,
                const float* __restrict__ Wout, const float* __restrict__ bout,
                const __bf16* __restrict__ Wpk0, const __bf16* __restrict__ Wpk1,
                __bf16* __restrict__ h0bf, __bf16* __restrict__ h1bf,
                float* __restrict__ c0, float* __restrict__ c1,
                const float* __restrict__ bias0, const float* __restrict__ bias1,
                const float* __restrict__ wih0c,
                float* __restrict__ xin, unsigned* __restrict__ syncb,
                float* __restrict__ dout) {
  __shared__ __bf16 bstage[2][4][4][512];  // [buf][kk][gate][frag] = 2x16KB
  __shared__ float gl[4][64][16];          // [gate][batch row][unit] staging
  __shared__ float hred[64][16];           // out-projection partials
  __shared__ float lbuf[64];               // loss reduction (block 0)

  const int tid  = threadIdx.x;
  const int wv   = tid >> 5;
  const int lane = tid & 31;
  const int mt   = wv & 3;            // M tile (16 batch rows)
  const int gt0  = (wv >> 2) * 2;     // two gate tiles per wave
  const int gt1  = gt0 + 1;
  const int blk  = blockIdx.x;        // owns units [16*blk, 16*blk+16)
  const int mrow = mt * 16 + (lane & 15);
  const int koff = (lane >> 4) * 8;   // A-fragment K sub-offset per ISA layout
  const int sgt   = tid >> 6;         // staging: gate handled by this thread
  const int spart = tid & 63;         // staging: 16B piece within fragment
  const float bo = bout[0];
  float lossacc = 0.0f;

  // loop-invariant staging source bases (fragment streams for this block)
  const __bf16* wsrc0 = Wpk0 + ((size_t)(sgt * 64 + blk) * 32) * 512 + (size_t)spart * 8;
  const __bf16* wsrc1 = Wpk1 + ((size_t)(sgt * 64 + blk) * 64) * 512 + (size_t)spart * 8;

  for (int t = 0; t < TSTEPS; ++t) {
    const int cur = t & 1, prev = cur ^ 1;
    const float* xcur = xin + cur * 64;
    float* xnext = xin + prev * 64;

    if (blk == 0 && tid < 64) xnext[tid] = bo;  // out accumulator starts at b_out

    // ================= phase 1: g0 = h0_prev @ W_hh0^T (K=1024) =================
    {
      v8f acc0 = {}; v8f acc1 = {};
      const __bf16* hp = h0bf + (size_t)prev * BATCH * HDIM + (size_t)mrow * HDIM + koff;
      // preload chunk 0 (4 k-tiles of B) into LDS via async DMA
      #pragma unroll
      for (int kk = 0; kk < 4; ++kk)
        stage_issue(wsrc0 + (size_t)kk * 512, &bstage[0][kk][sgt][spart * 8]);
      async_wait0();
      __syncthreads();
      int cb = 0;
      for (int c = 0; c < 8; ++c) {
        if (c + 1 < 8) {
          const __bf16* ws = wsrc0 + (size_t)(c + 1) * 2048;
          #pragma unroll
          for (int kk = 0; kk < 4; ++kk)
            stage_issue(ws + (size_t)kk * 512, &bstage[cb ^ 1][kk][sgt][spart * 8]);
        }
        #pragma unroll
        for (int kk = 0; kk < 4; ++kk) {
          int kt = c * 4 + kk;
          v8bf lo = *(const v8bf*)(hp + kt * 32);
          v8bf hi = *(const v8bf*)(hp + kt * 32 + 16);
          v16bf a = __builtin_shufflevector(lo, hi, 0,1,2,3,4,5,6,7,8,9,10,11,12,13,14,15);
          v16bf b0 = *(const v16bf*)(&bstage[cb][kk][gt0][lane * 16]);
          v16bf b1 = *(const v16bf*)(&bstage[cb][kk][gt1][lane * 16]);
          acc0 = __builtin_amdgcn_wmma_f32_16x16x32_bf16(false, a, false, b0, (short)0, acc0, false, false);
          acc1 = __builtin_amdgcn_wmma_f32_16x16x32_bf16(false, a, false, b1, (short)0, acc1, false, false);
        }
        async_wait0();     // next chunk's DMA has landed (overlapped with WMMAs)
        __syncthreads();
        cb ^= 1;
      }
      const int mo = (lane >> 4) * 8, nn = lane & 15;
      #pragma unroll
      for (int r = 0; r < 8; ++r) {
        gl[gt0][mt * 16 + mo + r][nn] = acc0[r];
        gl[gt1][mt * 16 + mo + r][nn] = acc1[r];
      }
    }
    __syncthreads();
    // LSTM cell, layer 0 (adds x*W_ih0 column + biases)
    #pragma unroll
    for (int e = 0; e < 4; ++e) {
      int idx = tid + e * TPB;
      int m = idx >> 4, u = idx & 15;
      int ug = blk * 16 + u;
      float xi = xcur[m];
      float iv = gl[0][m][u] + bias0[ug]          + xi * wih0c[ug];
      float fv = gl[1][m][u] + bias0[1024 + ug]   + xi * wih0c[1024 + ug];
      float gv = gl[2][m][u] + bias0[2048 + ug]   + xi * wih0c[2048 + ug];
      float ov = gl[3][m][u] + bias0[3072 + ug]   + xi * wih0c[3072 + ug];
      float cc = c0[(size_t)m * HDIM + ug];
      float cn = sigf(fv) * cc + sigf(iv) * tanh_fast(gv);
      float hn = sigf(ov) * tanh_fast(cn);
      c0[(size_t)m * HDIM + ug] = cn;
      h0bf[(size_t)cur * BATCH * HDIM + (size_t)m * HDIM + ug] = (__bf16)hn;
    }
    gsync(syncb, syncb + 1);

    // ====== phase 2: g1 = [h0_cur ; h1_prev] @ [W_ih1 ; W_hh1]^T (K=2048) ======
    {
      v8f acc0 = {}; v8f acc1 = {};
      const __bf16* h0c = h0bf + (size_t)cur  * BATCH * HDIM + (size_t)mrow * HDIM + koff;
      const __bf16* h1p = h1bf + (size_t)prev * BATCH * HDIM + (size_t)mrow * HDIM + koff;
      #pragma unroll
      for (int kk = 0; kk < 4; ++kk)
        stage_issue(wsrc1 + (size_t)kk * 512, &bstage[0][kk][sgt][spart * 8]);
      async_wait0();
      __syncthreads();
      int cb = 0;
      for (int c = 0; c < 16; ++c) {
        if (c + 1 < 16) {
          const __bf16* ws = wsrc1 + (size_t)(c + 1) * 2048;
          #pragma unroll
          for (int kk = 0; kk < 4; ++kk)
            stage_issue(ws + (size_t)kk * 512, &bstage[cb ^ 1][kk][sgt][spart * 8]);
        }
        const __bf16* ab = (c < 8) ? h0c : h1p;     // uniform per chunk
        const int kb = (c < 8) ? c * 4 : (c - 8) * 4;
        #pragma unroll
        for (int kk = 0; kk < 4; ++kk) {
          v8bf lo = *(const v8bf*)(ab + (kb + kk) * 32);
          v8bf hi = *(const v8bf*)(ab + (kb + kk) * 32 + 16);
          v16bf a = __builtin_shufflevector(lo, hi, 0,1,2,3,4,5,6,7,8,9,10,11,12,13,14,15);
          v16bf b0 = *(const v16bf*)(&bstage[cb][kk][gt0][lane * 16]);
          v16bf b1 = *(const v16bf*)(&bstage[cb][kk][gt1][lane * 16]);
          acc0 = __builtin_amdgcn_wmma_f32_16x16x32_bf16(false, a, false, b0, (short)0, acc0, false, false);
          acc1 = __builtin_amdgcn_wmma_f32_16x16x32_bf16(false, a, false, b1, (short)0, acc1, false, false);
        }
        async_wait0();
        __syncthreads();
        cb ^= 1;
      }
      const int mo = (lane >> 4) * 8, nn = lane & 15;
      #pragma unroll
      for (int r = 0; r < 8; ++r) {
        gl[gt0][mt * 16 + mo + r][nn] = acc0[r];
        gl[gt1][mt * 16 + mo + r][nn] = acc1[r];
      }
    }
    __syncthreads();
    // LSTM cell, layer 1 + output-projection partials
    #pragma unroll
    for (int e = 0; e < 4; ++e) {
      int idx = tid + e * TPB;
      int m = idx >> 4, u = idx & 15;
      int ug = blk * 16 + u;
      float iv = gl[0][m][u] + bias1[ug];
      float fv = gl[1][m][u] + bias1[1024 + ug];
      float gv = gl[2][m][u] + bias1[2048 + ug];
      float ov = gl[3][m][u] + bias1[3072 + ug];
      float cc = c1[(size_t)m * HDIM + ug];
      float cn = sigf(fv) * cc + sigf(iv) * tanh_fast(gv);
      float hn = sigf(ov) * tanh_fast(cn);
      c1[(size_t)m * HDIM + ug] = cn;
      h1bf[(size_t)cur * BATCH * HDIM + (size_t)m * HDIM + ug] = (__bf16)hn;
      hred[m][u] = hn * Wout[ug];
    }
    __syncthreads();
    if (tid < 64) {
      float s = 0.0f;
      #pragma unroll
      for (int u = 0; u < 16; ++u) s += hred[tid][u];
      atomicAdd(&xnext[tid], s);      // out[m] = b_out + sum partials
    }
    gsync(syncb, syncb + 1);

    // block 0: emit outs[:, t] and accumulate step MSE loss
    if (blk == 0) {
      if (tid < 64) {
        float o = xnext[tid];
        dout[1 + (size_t)tid * TSTEPS + t] = o;
        float d = o - inputs[(size_t)tid * TSTEPS + t];
        lbuf[tid] = d * d;
      }
      __syncthreads();
      if (tid == 0) {
        float s = 0.0f;
        for (int m = 0; m < 64; ++m) s += lbuf[m];
        lossacc += s * (1.0f / 64.0f);
      }
      __syncthreads();
    }
  }
  if (blk == 0 && tid == 0) dout[0] = lossacc;
}

extern "C" void kernel_launch(void* const* d_in, const int* in_sizes, int n_in,
                              void* d_out, int out_size, void* d_ws, size_t ws_size,
                              hipStream_t stream) {
  (void)in_sizes; (void)n_in; (void)out_size; (void)ws_size;
  const float* inputs = (const float*)d_in[0];
  const float* z      = (const float*)d_in[1];
  const float* W_ih0  = (const float*)d_in[2];
  const float* W_hh0  = (const float*)d_in[3];
  const float* b_ih0  = (const float*)d_in[4];
  const float* b_hh0  = (const float*)d_in[5];
  const float* W_ih1  = (const float*)d_in[6];
  const float* W_hh1  = (const float*)d_in[7];
  const float* b_ih1  = (const float*)d_in[8];
  const float* b_hh1  = (const float*)d_in[9];
  const float* W_out  = (const float*)d_in[10];
  const float* b_out  = (const float*)d_in[11];
  float* out = (float*)d_out;

  char* ws = (char*)d_ws;
  __bf16*   Wpk0  = (__bf16*)(ws + OFF_WPK0);
  __bf16*   Wpk1  = (__bf16*)(ws + OFF_WPK1);
  __bf16*   h0bf  = (__bf16*)(ws + OFF_H0BF);
  __bf16*   h1bf  = (__bf16*)(ws + OFF_H1BF);
  float*    c0    = (float*)(ws + OFF_C0);
  float*    c1    = (float*)(ws + OFF_C1);
  float*    bias0 = (float*)(ws + OFF_BIAS0);
  float*    bias1 = (float*)(ws + OFF_BIAS1);
  float*    wih0c = (float*)(ws + OFF_WIH0);
  float*    xinb  = (float*)(ws + OFF_XIN);
  unsigned* syncb = (unsigned*)(ws + OFF_SYNC);

  pack_w0<<<WPK0_ELEMS / TPB, TPB, 0, stream>>>(W_hh0, Wpk0);
  pack_w1<<<WPK1_ELEMS / TPB, TPB, 0, stream>>>(W_ih1, W_hh1, Wpk1);
  init_state<<<(BATCH * HDIM) / TPB, TPB, 0, stream>>>(
      z, b_ih0, b_hh0, b_ih1, b_hh1, W_ih0,
      c0, c1, h0bf, h1bf, bias0, bias1, wih0c, xinb, syncb);
  lstm_persistent<<<NBLK, TPB, 0, stream>>>(
      inputs, W_out, b_out, Wpk0, Wpk1, h0bf, h1bf, c0, c1,
      bias0, bias1, wih0c, xinb, syncb, out);
}
